// ExpansionContrastModule_8031588844294
// MI455X (gfx1250) — matrix operand order repaired and checked by reading, not compiled, not run on hardware
//
#include <hip/hip_runtime.h>
#include <hip/hip_bf16.h>
#include <math.h>

// ============================================================================
// ExpansionContrastModule for MI455X (gfx1250, wave32, WMMA bf16).
// All heavy contractions go through v_wmma_f32_16x16x32_bf16.
// Workspace requirement: ~76 MB (buffers aliased; see partition below).
// ============================================================================

typedef __attribute__((ext_vector_type(16))) __bf16 bf16x16;
typedef __attribute__((ext_vector_type(8)))  float  f32x8;

union FragAB { bf16x16 v; uint4 q[2]; };

__device__ __forceinline__ unsigned short f2bf(float f) {
  union { float f; unsigned u; } x; x.f = f;
  unsigned r = x.u + 0x7FFFu + ((x.u >> 16) & 1u);   // round-to-nearest-even
  return (unsigned short)(r >> 16);
}

// ----------------------------------------------------------------------------
// Generic NT GEMM: C[M x N] = A[M x K] * B[N x K]^T, A/B bf16 (row-major,
// K contiguous), accumulate f32. MODE 0: store f32 C (stride ldc).
// MODE 1: store bf16 C^T at Ct[n*ldc + m]. MODE 2: C[m*ldc+n] -= result.
// Requires M,N multiples of 64; K multiple of 32; lda/ldb multiples of 8.
// Block = 128 threads (4 wave32), each wave does a 32x32 tile (2x2 WMMA).
// ----------------------------------------------------------------------------
template<int MODE>
__global__ __launch_bounds__(128)
void gemm_nt_bf16(const unsigned short* __restrict__ A,
                  const unsigned short* __restrict__ B,
                  float* __restrict__ C, unsigned short* __restrict__ Ct,
                  int M, int N, int K, int lda, int ldb, int ldc) {
  (void)M;
  __shared__ unsigned short As[64 * 40];   // 64 rows x 32 k, padded to 40
  __shared__ unsigned short Bs[64 * 40];

  const int bm = blockIdx.y * 64;
  const int bn = blockIdx.x * 64;
  const int t    = threadIdx.x;
  const int wave = t >> 5;
  const int lane = t & 31;
  const int wm = (wave >> 1) * 32;     // wave's 32x32 sub-tile
  const int wn = (wave & 1) * 32;
  const int l16   = lane & 15;
  const bool loHalf = (lane < 16);

  const int ldRow  = t >> 1;           // cooperative-load row (0..63)
  const int ldHalf = t & 1;            // k-chunk 0..15 / 16..31

  f32x8 acc[2][2];
  for (int i = 0; i < 2; ++i)
    for (int j = 0; j < 2; ++j)
      for (int e = 0; e < 8; ++e) acc[i][j][e] = 0.0f;

  for (int k0 = 0; k0 < K; k0 += 32) {
    __syncthreads();
    // ---- cooperative global->LDS (32 B per thread per matrix, b128 loads)
    {
      const uint4* ga = (const uint4*)(A + (size_t)(bm + ldRow) * lda + k0 + ldHalf * 16);
      uint4 a0 = ga[0], a1 = ga[1];
      *(uint4*)&As[ldRow * 40 + ldHalf * 16]     = a0;
      *(uint4*)&As[ldRow * 40 + ldHalf * 16 + 8] = a1;
      const uint4* gb = (const uint4*)(B + (size_t)(bn + ldRow) * ldb + k0 + ldHalf * 16);
      uint4 b0 = gb[0], b1 = gb[1];
      *(uint4*)&Bs[ldRow * 40 + ldHalf * 16]     = b0;
      *(uint4*)&Bs[ldRow * 40 + ldHalf * 16 + 8] = b1;
    }
    __syncthreads();

    // ---- fragment reads (ds_load_b128) per CDNA5 16-bit A/B VGPR layouts
    FragAB a[2], bf[2];
    const int aLo = loHalf ? 0 : 8;      // A: lanes 0-15 K={0..7,16..23}
    const int bLo = loHalf ? 0 : 8;      // B: lanes 0-15 K={0..7,16..23}
    for (int rr = 0; rr < 2; ++rr) {
      const int m = wm + rr * 16 + l16;
      a[rr].q[0] = *(const uint4*)&As[m * 40 + aLo];
      a[rr].q[1] = *(const uint4*)&As[m * 40 + aLo + 16];
    }
    for (int cc = 0; cc < 2; ++cc) {
      const int n = wn + cc * 16 + l16;
      bf[cc].q[0] = *(const uint4*)&Bs[n * 40 + bLo];
      bf[cc].q[1] = *(const uint4*)&Bs[n * 40 + bLo + 16];
    }

    for (int rr = 0; rr < 2; ++rr)
      for (int cc = 0; cc < 2; ++cc)
        acc[rr][cc] = __builtin_amdgcn_wmma_f32_16x16x32_bf16(
            false, a[rr].v, false, bf[cc].v, (short)0, acc[rr][cc], false, false);
  }

  // ---- epilogue: C/D layout: VGPR j -> M = j (lanes 0-15) / j+8 (16-31)
  for (int rr = 0; rr < 2; ++rr)
    for (int cc = 0; cc < 2; ++cc)
      for (int j = 0; j < 8; ++j) {
        const int row = bm + wm + rr * 16 + (loHalf ? j : j + 8);
        const int col = bn + wn + cc * 16 + l16;
        const float v = acc[rr][cc][j];
        if (MODE == 0) {
          C[(size_t)row * ldc + col] = v;
        } else if (MODE == 1) {
          Ct[(size_t)col * ldc + row] = f2bf(v);
        } else {  // MODE 2: subtract-accumulate
          float* cp = &C[(size_t)row * ldc + col];
          *cp = *cp - v;
        }
      }
}

// ----------------------------------------------------------------------------
// f32 -> bf16 flat convert (weights)
// ----------------------------------------------------------------------------
__global__ void cvt_f32_bf16(const float* __restrict__ in,
                             unsigned short* __restrict__ out, int n) {
  int i = blockIdx.x * 256 + threadIdx.x;
  if (i < n) out[i] = f2bf(in[i]);
}

// ----------------------------------------------------------------------------
// Tiled transpose: in [R x Ccol] f32 -> out [Ccol x R] bf16
// ----------------------------------------------------------------------------
__global__ __launch_bounds__(256)
void transpose_f32_bf16(const float* __restrict__ in,
                        unsigned short* __restrict__ out, int R, int Ccol) {
  __shared__ float tile[32][33];
  const int cb = blockIdx.x * 32, rb = blockIdx.y * 32;
  const int tx = threadIdx.x, ty = threadIdx.y;     // (32, 8)
  for (int i = 0; i < 32; i += 8)
    tile[ty + i][tx] = in[(size_t)(rb + ty + i) * Ccol + cb + tx];
  __syncthreads();
  for (int i = 0; i < 32; i += 8)
    out[(size_t)(cb + ty + i) * R + rb + tx] = f2bf(tile[tx][ty + i]);
}

// ----------------------------------------------------------------------------
// Build transposed neighbor tensor: surrT[a][n*256+c] = cenT[shift_n(a)][c]
// grid (4096, 8), block 128 (each thread moves 2 bf16 = 1 dword)
// ----------------------------------------------------------------------------
__global__ __launch_bounds__(128)
void build_surrT(const unsigned short* __restrict__ cenT,
                 unsigned short* __restrict__ surrT, int d) {
  const int dy[8] = {-1, -1, -1, 0, 1, 1, 1, 0};
  const int dx[8] = {-1, 0, 1, 1, 1, 0, -1, -1};
  const int a = blockIdx.x, n = blockIdx.y, t = threadIdx.x;
  const int y = a >> 6, x = a & 63;
  const int ys = y + dy[n] * d, xs = x + dx[n] * d;
  unsigned v = 0u;
  if ((unsigned)ys < 64u && (unsigned)xs < 64u)
    v = ((const unsigned*)(cenT + (size_t)(ys * 64 + xs) * 256))[t];
  ((unsigned*)(surrT + (size_t)a * 2048 + n * 256))[t] = v;
}

// ----------------------------------------------------------------------------
// Row-wise L2 normalize: out_bf16[r][c] = in[r][c] / max(||row||, 1e-12)
// ----------------------------------------------------------------------------
__global__ __launch_bounds__(256)
void normalize_rows_bf16(const float* __restrict__ in,
                         unsigned short* __restrict__ out, int ncols) {
  __shared__ float red[256];
  const int r = blockIdx.x, t = threadIdx.x;
  const float* row = in + (size_t)r * ncols;
  float s = 0.f;
  for (int c = t; c < ncols; c += 256) { float v = row[c]; s += v * v; }
  red[t] = s; __syncthreads();
  for (int o = 128; o; o >>= 1) { if (t < o) red[t] += red[t + o]; __syncthreads(); }
  const float inv = 1.0f / fmaxf(sqrtf(red[0]), 1e-12f);
  for (int c = t; c < ncols; c += 256)
    out[(size_t)r * ncols + c] = f2bf(row[c] * inv);
}

// ----------------------------------------------------------------------------
// InstanceNorm over [64x512] + row softmax + wc = sum of 8 chunks.
// One block, 512 threads.
// ----------------------------------------------------------------------------
__global__ __launch_bounds__(512)
void attn_post(const float* __restrict__ score,
               unsigned short* __restrict__ wsb,
               unsigned short* __restrict__ wcb) {
  __shared__ float red[512];
  __shared__ float sWc[64];
  __shared__ float sMu, sRstd, sMax, sSum;
  const int t = threadIdx.x;

  float s = 0.f, s2 = 0.f;
  for (int i = t; i < 64 * 512; i += 512) { float v = score[i]; s += v; s2 += v * v; }
  red[t] = s; __syncthreads();
  for (int o = 256; o; o >>= 1) { if (t < o) red[t] += red[t + o]; __syncthreads(); }
  if (t == 0) sMu = red[0] * (1.f / 32768.f);
  __syncthreads();
  red[t] = s2; __syncthreads();
  for (int o = 256; o; o >>= 1) { if (t < o) red[t] += red[t + o]; __syncthreads(); }
  if (t == 0) {
    float m = sMu;
    float var = red[0] * (1.f / 32768.f) - m * m;
    sRstd = rsqrtf(var + 1e-5f);
  }
  __syncthreads();
  const float mu = sMu, rstd = sRstd;

  for (int r = 0; r < 64; ++r) {
    const float x = (score[r * 512 + t] - mu) * rstd;
    red[t] = x; __syncthreads();
    for (int o = 256; o; o >>= 1) { if (t < o) red[t] = fmaxf(red[t], red[t + o]); __syncthreads(); }
    if (t == 0) sMax = red[0];
    __syncthreads();
    const float e = __expf(x - sMax);
    red[t] = e; __syncthreads();
    for (int o = 256; o; o >>= 1) { if (t < o) red[t] += red[t + o]; __syncthreads(); }
    if (t == 0) sSum = red[0];
    if (t < 64) sWc[t] = 0.f;
    __syncthreads();
    const float w = e / sSum;
    wsb[r * 512 + t] = f2bf(w);
    atomicAdd(&sWc[t & 63], w);
    __syncthreads();
    if (t < 64) wcb[r * 64 + t] = f2bf(sWc[t]);
    __syncthreads();
  }
}

// ----------------------------------------------------------------------------
// BatchNorm (training) stats per channel over (b, area): z [4][256][4096]
// ----------------------------------------------------------------------------
__global__ __launch_bounds__(256)
void bn_stats(const float* __restrict__ z, float* __restrict__ mean,
              float* __restrict__ rstd) {
  __shared__ float r1[256], r2[256];
  const int o = blockIdx.x, t = threadIdx.x;
  float s = 0.f, s2 = 0.f;
  for (int b = 0; b < 4; ++b) {
    const float* p = z + ((size_t)b * 256 + o) * 4096;
    for (int a = t; a < 4096; a += 256) { float v = p[a]; s += v; s2 += v * v; }
  }
  r1[t] = s; r2[t] = s2; __syncthreads();
  for (int off = 128; off; off >>= 1) {
    if (t < off) { r1[t] += r1[t + off]; r2[t] += r2[t + off]; }
    __syncthreads();
  }
  if (t == 0) {
    float m = r1[0] * (1.f / 16384.f);
    float var = r2[0] * (1.f / 16384.f) - m * m;
    mean[o] = m;
    rstd[o] = rsqrtf(var + 1e-5f);
  }
}

__global__ void bn_apply_relu(const float* __restrict__ z,
                              const float* __restrict__ mean,
                              const float* __restrict__ rstd,
                              const float* __restrict__ gamma,
                              const float* __restrict__ beta,
                              float* __restrict__ out, int total) {
  int i = blockIdx.x * 256 + threadIdx.x;
  if (i >= total) return;
  int o = (i >> 12) & 255;
  float v = (z[i] - mean[o]) * rstd[o] * gamma[o] + beta[o];
  out[i] = fmaxf(v, 0.0f);
}

// ============================================================================
// Host orchestration
// ============================================================================
extern "C" void kernel_launch(void* const* d_in, const int* in_sizes, int n_in,
                              void* d_out, int out_size, void* d_ws, size_t ws_size,
                              hipStream_t stream) {
  (void)in_sizes; (void)n_in; (void)out_size; (void)ws_size;
  const float* cen   = (const float*)d_in[0];  // [4,256,64,64]
  const float* qw    = (const float*)d_in[1];  // [4,64,256]
  const float* lw    = (const float*)d_in[2];  // [4,64,256]
  const float* kw    = (const float*)d_in[3];  // [4,512,2048]
  const float* vw    = (const float*)d_in[4];  // [4,512,2048]
  const float* ow    = (const float*)d_in[5];  // [256,256]
  const float* gamma = (const float*)d_in[6];  // [256]
  const float* beta  = (const float*)d_in[7];  // [256]
  float* out = (float*)d_out;                  // [4,256,64,64]

  // ---- workspace partition (256B aligned). Total ~76 MB.
  char* p = (char*)d_ws;
  auto alloc = [&](size_t bytes) -> char* {
    char* r = p; p += (bytes + 255) & ~(size_t)255; return r;
  };
  unsigned short* qwb     = (unsigned short*)alloc((size_t)4*64*256*2);
  unsigned short* lwb     = (unsigned short*)alloc((size_t)4*64*256*2);
  unsigned short* kwb     = (unsigned short*)alloc((size_t)4*512*2048*2);
  unsigned short* vwb     = (unsigned short*)alloc((size_t)4*512*2048*2);
  unsigned short* owb     = (unsigned short*)alloc((size_t)256*256*2);
  unsigned short* cenT    = (unsigned short*)alloc((size_t)4*4096*256*2);  // also outpreT later
  unsigned short* surrT   = (unsigned short*)alloc((size_t)4096*2048*2);   // also zbuf later
  float*          keys    = (float*)         alloc((size_t)512*4096*4);
  unsigned short* kn      = (unsigned short*)alloc((size_t)512*4096*2);
  unsigned short* valsT   = (unsigned short*)alloc((size_t)4096*512*2);
  float*          qs      = (float*)         alloc((size_t)64*4096*4);
  unsigned short* qn      = (unsigned short*)alloc((size_t)64*4096*2);
  unsigned short* locsT   = (unsigned short*)alloc((size_t)4096*64*2);
  float*          score   = (float*)         alloc((size_t)64*512*4);
  unsigned short* wsb     = (unsigned short*)alloc((size_t)64*512*2);
  unsigned short* wcb     = (unsigned short*)alloc((size_t)64*64*2);
  float*          outpre  = (float*)         alloc((size_t)4*256*4096*4);
  float*          bnMean  = (float*)         alloc((size_t)256*4);
  float*          bnRstd  = (float*)         alloc((size_t)256*4);
  float*          zbuf    = (float*)surrT;            // alias (surrT dead by phase 4)
  unsigned short* outpreT = cenT;                      // alias (cenT dead by phase 4)

  // ---- phase 0: weights -> bf16
  auto cvt = [&](const float* src, unsigned short* dst, int n) {
    cvt_f32_bf16<<<(n + 255) / 256, 256, 0, stream>>>(src, dst, n);
  };
  cvt(qw, qwb, 4*64*256);
  cvt(lw, lwb, 4*64*256);
  cvt(kw, kwb, 4*512*2048);
  cvt(vw, vwb, 4*512*2048);
  cvt(ow, owb, 256*256);

  // ---- phase 1: cenT[b] = cen[b]^T  ([256 x 4096] -> [4096 x 256] bf16)
  for (int b = 0; b < 4; ++b)
    transpose_f32_bf16<<<dim3(4096/32, 256/32), dim3(32, 8), 0, stream>>>(
        cen + (size_t)b * 256 * 4096, cenT + (size_t)b * 4096 * 256, 256, 4096);

  // ---- phase 2: per (b, s) attention pipeline
  for (int b = 0; b < 4; ++b) {
    const unsigned short* cenTb = cenT + (size_t)b * 4096 * 256;
    for (int s = 0; s < 4; ++s) {
      const int d = s + 1;
      // qs = query_w[s] @ cen[b]   -> f32 [64 x 4096]
      gemm_nt_bf16<0><<<dim3(64, 1), 128, 0, stream>>>(
          qwb + (size_t)s*64*256, cenTb, qs, nullptr, 64, 4096, 256, 256, 256, 4096);
      normalize_rows_bf16<<<64, 256, 0, stream>>>(qs, qn, 4096);
      // locsT = (local_w[s] @ cen[b])^T -> bf16 [4096 x 64]
      gemm_nt_bf16<1><<<dim3(64, 1), 128, 0, stream>>>(
          lwb + (size_t)s*64*256, cenTb, nullptr, locsT, 64, 4096, 256, 256, 256, 64);
      // surrT = 8 dilated-shift neighbors, transposed  [4096 x 2048] bf16
      build_surrT<<<dim3(4096, 8), 128, 0, stream>>>(cenTb, surrT, d);
      // keys = key_w[s] @ surr -> f32 [512 x 4096]
      gemm_nt_bf16<0><<<dim3(64, 8), 128, 0, stream>>>(
          kwb + (size_t)s*512*2048, surrT, keys, nullptr, 512, 4096, 2048, 2048, 2048, 4096);
      // valsT = (value_w[s] @ surr)^T -> bf16 [4096 x 512]
      gemm_nt_bf16<1><<<dim3(64, 8), 128, 0, stream>>>(
          vwb + (size_t)s*512*2048, surrT, nullptr, valsT, 512, 4096, 2048, 2048, 2048, 512);
      normalize_rows_bf16<<<512, 256, 0, stream>>>(keys, kn, 4096);
      // score = qn @ kn^T -> f32 [64 x 512]
      gemm_nt_bf16<0><<<dim3(8, 1), 128, 0, stream>>>(
          qn, kn, score, nullptr, 64, 512, 4096, 4096, 4096, 512);
      // instance-norm + softmax + chunk-sum
      attn_post<<<1, 512, 0, stream>>>(score, wsb, wcb);
      // outpre[b, s*64:(s+1)*64, :] = ws @ vals - wc @ locs
      float* outSlice = outpre + ((size_t)b * 256 + s * 64) * 4096;
      gemm_nt_bf16<0><<<dim3(64, 1), 128, 0, stream>>>(
          wsb, valsT, outSlice, nullptr, 64, 4096, 512, 512, 512, 4096);
      gemm_nt_bf16<2><<<dim3(64, 1), 128, 0, stream>>>(
          wcb, locsT, outSlice, nullptr, 64, 4096, 64, 64, 64, 4096);
    }
  }

  // ---- phase 3: out_conv (1x1) via GEMM per batch
  for (int b = 0; b < 4; ++b) {
    transpose_f32_bf16<<<dim3(4096/32, 256/32), dim3(32, 8), 0, stream>>>(
        outpre + (size_t)b * 256 * 4096, outpreT, 256, 4096);
    gemm_nt_bf16<0><<<dim3(64, 4), 128, 0, stream>>>(
        owb, outpreT, zbuf + (size_t)b * 256 * 4096, nullptr,
        256, 4096, 256, 256, 256, 4096);
  }

  // ---- phase 4: BatchNorm (training stats) + ReLU
  bn_stats<<<256, 256, 0, stream>>>(zbuf, bnMean, bnRstd);
  const int total = 4 * 256 * 4096;
  bn_apply_relu<<<total / 256, 256, 0, stream>>>(zbuf, bnMean, bnRstd, gamma, beta, out, total);
}
